// InterModalityUpdate_86260123173714
// MI455X (gfx1250) — matrix operand order repaired and checked by reading, not compiled, not run on hardware
//
#include <hip/hip_runtime.h>
#include <cstdint>

// ---------------------------------------------------------------------------
// Problem constants (from reference): B=64, NO=256, ML=128, VS=2048, QS=1024,
// OUT=2048, H=16, dh=128, C6=3*OUT=6144.
// ---------------------------------------------------------------------------
#define NB   64
#define NO_  256
#define ML_  128
#define OUT_ 2048
#define C6_  6144
#define DH_  128

typedef __attribute__((ext_vector_type(16))) __bf16 v16bf;
typedef __attribute__((ext_vector_type(8)))  float  v8f;

union frag32 { uint4 u[2]; v16bf v; };          // 32 bytes = 16 bf16
union h16frag { uint16_t h[16]; v16bf v; };

__device__ __forceinline__ uint16_t f2bf(float f) {
    union { float f; uint32_t u; } x; x.f = f;
    uint32_t u = x.u;
    uint32_t r = u + 0x7FFFu + ((u >> 16) & 1u);   // round-to-nearest-even
    return (uint16_t)(r >> 16);
}

__device__ __forceinline__ v8f wmma_bf16(v16bf a, v16bf b, v8f c) {
    return __builtin_amdgcn_wmma_f32_16x16x32_bf16(
        /*neg_a=*/false, a, /*neg_b=*/false, b,
        /*c_mod=*/(short)0, c, /*reuse_a=*/false, /*reuse_b=*/false);
}

// --- CDNA5 async memory->LDS path (ASYNCcnt-tracked, no VGPR data) ---------
// LDS byte offset = low 32 bits of the flat address of a __shared__ pointer
// (ISA aperture mapping).
__device__ __forceinline__ uint32_t lds_off(const void* p) {
    return (uint32_t)(uintptr_t)p;
}
__device__ __forceinline__ void async_load_b128(uint32_t lds, const void* g) {
    asm volatile("global_load_async_to_lds_b128 %0, %1, off"
                 :: "v"(lds), "v"(g) : "memory");
}
__device__ __forceinline__ void wait_async0() {
    asm volatile("s_wait_asynccnt 0" ::: "memory");
}

// ---------------------------------------------------------------------------
// f32 -> bf16 bulk conversion (4 elements / thread, all sizes % 1024 == 0)
// ---------------------------------------------------------------------------
__global__ __launch_bounds__(256) void cvt_f32_bf16(
        const float* __restrict__ in, uint16_t* __restrict__ out, long long n) {
    long long i = ((long long)blockIdx.x * 256 + threadIdx.x) * 4;
    if (i >= n) return;                    // grids are exact
    float4 v = *(const float4*)(in + i);
    ushort4 o;
    o.x = f2bf(v.x); o.y = f2bf(v.y); o.z = f2bf(v.z); o.w = f2bf(v.w);
    *(ushort4*)(out + i) = o;
}

// ---------------------------------------------------------------------------
// Generic bf16 WMMA GEMM:  C[M,N] = concat_k(A1,A2)[M,K] . W[N,K]^T + bias[N]
// Compile-time HAS_MASK (per-row mask multiply) and OUT_BF16 selection.
// Block tile 128x128, 256 threads = 8 waves (4 M-slots x 2 N-slots),
// each wave computes 32x64 = 2x4 WMMA 16x16 tiles. K step = 32.
// Double-buffered LDS tiles filled with global_load_async_to_lds_b128;
// the async engine overlaps the next tile's fill with this tile's WMMAs.
// LDS pitch 56 elements (112B): 16B ds_load_b128 fragment reads are
// bank-conflict-free across the 16 lanes of each half-wave.
// ---------------------------------------------------------------------------
#define GP 56
template <bool HAS_MASK, bool OUT_BF16>
__global__ __launch_bounds__(256) void gemm_wmma(
        const uint16_t* __restrict__ A1, const uint16_t* __restrict__ A2,
        int ksplit,
        const uint16_t* __restrict__ W, const float* __restrict__ bias,
        const float* __restrict__ mask, int M, int N, int K,
        uint16_t* __restrict__ outb, float* __restrict__ outf) {
    __shared__ uint16_t As[2][128 * GP];
    __shared__ uint16_t Ws[2][128 * GP];

    const int tid  = threadIdx.x;
    const int wave = tid >> 5, lane = tid & 31;
    const int wm = wave >> 1, wn = wave & 1;
    const int l15 = lane & 15, lhi = lane >> 4;
    const int bm = blockIdx.y * 128, bn = blockIdx.x * 128;

    const v8f vzero = {0.f,0.f,0.f,0.f,0.f,0.f,0.f,0.f};
    v8f acc[2][4];
#pragma unroll
    for (int i = 0; i < 2; ++i)
#pragma unroll
        for (int j = 0; j < 4; ++j) acc[i][j] = vzero;

    const int pitchA2 = K - ksplit;

    // per-thread staging slots: 2 x 16B chunks of A and of W per tile
    const int r0 = tid >> 2,          s0 = (tid & 3) * 8;          // chunk 0
    const int r1 = (tid + 256) >> 2,  s1 = s0;                     // chunk 1

    auto stage = [&](int k0, int buf) {
        const uint16_t* aSrc; int apitch, acol;
        if (k0 < ksplit) { aSrc = A1; apitch = ksplit;  acol = k0; }
        else             { aSrc = A2; apitch = pitchA2; acol = k0 - ksplit; }
        async_load_b128(lds_off(&As[buf][r0 * GP + s0]),
                        &aSrc[(size_t)(bm + r0) * apitch + acol + s0]);
        async_load_b128(lds_off(&As[buf][r1 * GP + s1]),
                        &aSrc[(size_t)(bm + r1) * apitch + acol + s1]);
        async_load_b128(lds_off(&Ws[buf][r0 * GP + s0]),
                        &W[(size_t)(bn + r0) * K + k0 + s0]);
        async_load_b128(lds_off(&Ws[buf][r1 * GP + s1]),
                        &W[(size_t)(bn + r1) * K + k0 + s1]);
    };

    const int nk = K >> 5;
    stage(0, 0);
    for (int kt = 0; kt < nk; ++kt) {
        wait_async0();            // my async fills of buf[kt&1] are done
        __syncthreads();          // everyone's fills done / prior reads done
        if (kt + 1 < nk) stage((kt + 1) << 5, (kt + 1) & 1);

        const uint16_t* Ab = As[kt & 1];
        const uint16_t* Wb = Ws[kt & 1];
        v16bf afrag[2], bfrag[4];
#pragma unroll
        for (int mi = 0; mi < 2; ++mi) {   // A 16x32: lanes 0-15 K 0-15
            int lr   = wm * 32 + mi * 16 + l15;
            int koff = lhi ? 16 : 0;
            frag32 f;
            f.u[0] = *(const uint4*)&Ab[lr * GP + koff];
            f.u[1] = *(const uint4*)&Ab[lr * GP + koff + 8];
            afrag[mi] = f.v;
        }
#pragma unroll
        for (int ni = 0; ni < 4; ++ni) {   // B 32x16: lanes 0-15 K 0-15
            int wr = wn * 64 + ni * 16 + l15;
            int kb = lhi ? 16 : 0;
            frag32 f;
            f.u[0] = *(const uint4*)&Wb[wr * GP + kb];
            f.u[1] = *(const uint4*)&Wb[wr * GP + kb + 8];
            bfrag[ni] = f.v;
        }
#pragma unroll
        for (int mi = 0; mi < 2; ++mi)
#pragma unroll
            for (int ni = 0; ni < 4; ++ni)
                acc[mi][ni] = wmma_bf16(afrag[mi], bfrag[ni], acc[mi][ni]);
    }

    // epilogue: lane holds col l15, rows r + 8*lhi of each 16x16 tile
    float mval[2];
    if (HAS_MASK) {
#pragma unroll
        for (int mi = 0; mi < 2; ++mi) (void)mi;  // masks loaded per row below
    }
    (void)mval;
#pragma unroll
    for (int mi = 0; mi < 2; ++mi) {
#pragma unroll
        for (int ni = 0; ni < 4; ++ni) {
            int gn = bn + wn * 64 + ni * 16 + l15;
            float bval = bias[gn];
#pragma unroll
            for (int r = 0; r < 8; ++r) {
                int gm = bm + wm * 32 + mi * 16 + lhi * 8 + r;
                float val = acc[mi][ni][r] + bval;
                if (HAS_MASK) val *= mask[gm];
                if (OUT_BF16) outb[(size_t)gm * N + gn] = f2bf(val);
                else          outf[(size_t)gm * N + gn] = val;
            }
        }
    }
}

// ---------------------------------------------------------------------------
// q->v attention: per (b,h), S = v_qry . q_key^T * scale, mask cols by
// q_mask, softmax over ML=128, then v_update = P . q_val. 8 waves, each
// handles 2 of the 16 v-row tiles. q_val is staged transposed in LDS so
// the P.V B-fragments are contiguous ds_load_b128.
// ---------------------------------------------------------------------------
__global__ __launch_bounds__(256) void attn_q2v(
        const uint16_t* __restrict__ vt, const uint16_t* __restrict__ qt,
        const float* __restrict__ qmask, uint16_t* __restrict__ vu) {
    const int h = blockIdx.x, b = blockIdx.y;
    const int tid = threadIdx.x, wave = tid >> 5, lane = tid & 31;
    const int l15 = lane & 15, lhi = lane >> 4;
    const float scale = 0.0883883476483184f;   // 1/sqrt(128)

    __shared__ uint16_t qvT[DH_ * ML_];        // q_val transposed [d][tok] 32KB
    __shared__ uint16_t pbuf[8 * 16 * ML_];    // per-wave P scratch     32KB

    for (int i = 0; i < 64; ++i) {             // stage q_val^T (coalesced read)
        int idx = tid + i * 256;               // 0..16383
        int tok = idx >> 7, d = idx & 127;
        qvT[d * ML_ + tok] =
            qt[((size_t)b * ML_ + tok) * C6_ + 2 * OUT_ + h * DH_ + d];
    }
    __syncthreads();

    uint16_t* pw = &pbuf[wave * 16 * ML_];
    float qm[8];
#pragma unroll
    for (int nt = 0; nt < 8; ++nt) qm[nt] = qmask[b * ML_ + nt * 16 + l15];

    const v8f vzero = {0.f,0.f,0.f,0.f,0.f,0.f,0.f,0.f};

    for (int t = wave; t < 16; t += 8) {
        const int vr0 = t * 16;
        v8f acc[8];
#pragma unroll
        for (int nt = 0; nt < 8; ++nt) acc[nt] = vzero;

        // ---- scores: S[16 x 128] over dh=128 (4 k-steps) ----
        for (int kk = 0; kk < 4; ++kk) {
            int k0 = kk * 32;
            frag32 fa;                          // A = v_qry rows (contiguous)
            {
                int vr = vr0 + l15, koff = lhi ? 16 : 0;
                const uint16_t* p =
                    vt + ((size_t)b * NO_ + vr) * C6_ + OUT_ + h * DH_ + k0 + koff;
                fa.u[0] = *(const uint4*)p;
                fa.u[1] = *(const uint4*)(p + 8);
            }
#pragma unroll
            for (int nt = 0; nt < 8; ++nt) {    // B = q_key cols (contiguous)
                frag32 fb;
                int tok = nt * 16 + l15, kb = lhi ? 16 : 0;
                const uint16_t* p =
                    qt + ((size_t)b * ML_ + tok) * C6_ + h * DH_ + k0 + kb;
                fb.u[0] = *(const uint4*)p;
                fb.u[1] = *(const uint4*)(p + 8);
                acc[nt] = wmma_bf16(fa.v, fb.v, acc[nt]);
            }
        }

        // ---- masked softmax over 128 cols (rows split across lane halves) --
#pragma unroll
        for (int r = 0; r < 8; ++r) {
            float mx = -3.0e38f;
#pragma unroll
            for (int nt = 0; nt < 8; ++nt) {
                float s = acc[nt][r] * scale;
                s = (qm[nt] == 0.0f) ? -1.0e30f : s;
                acc[nt][r] = s;
                mx = fmaxf(mx, s);
            }
#pragma unroll
            for (int xm = 1; xm < 16; xm <<= 1)
                mx = fmaxf(mx, __shfl_xor(mx, xm, 32));
            float sum = 0.f;
#pragma unroll
            for (int nt = 0; nt < 8; ++nt) {
                float e = __expf(acc[nt][r] - mx);
                acc[nt][r] = e;
                sum += e;
            }
#pragma unroll
            for (int xm = 1; xm < 16; xm <<= 1)
                sum += __shfl_xor(sum, xm, 32);
            float inv = 1.0f / sum;
            int pr = lhi * 8 + r;
#pragma unroll
            for (int nt = 0; nt < 8; ++nt)
                pw[pr * ML_ + nt * 16 + l15] = f2bf(acc[nt][r] * inv);
        }

        // ---- v_update tile = P(16x128) . q_val(128x128) ----
        v8f acc2[8];
#pragma unroll
        for (int dt = 0; dt < 8; ++dt) acc2[dt] = vzero;
        for (int mm = 0; mm < 4; ++mm) {
            int m0 = mm * 32;
            frag32 fa;                          // A = P from LDS scratch
            {
                int koff = lhi ? 16 : 0;
                const uint16_t* p = pw + l15 * ML_ + m0 + koff;
                fa.u[0] = *(const uint4*)p;
                fa.u[1] = *(const uint4*)(p + 8);
            }
#pragma unroll
            for (int dt = 0; dt < 8; ++dt) {    // B = q_val^T rows from LDS
                frag32 fb;
                int d = dt * 16 + l15, kb = lhi ? 16 : 0;
                const uint16_t* p = &qvT[d * ML_ + m0 + kb];
                fb.u[0] = *(const uint4*)p;
                fb.u[1] = *(const uint4*)(p + 8);
                acc2[dt] = wmma_bf16(fa.v, fb.v, acc2[dt]);
            }
        }
#pragma unroll
        for (int dt = 0; dt < 8; ++dt)
#pragma unroll
            for (int r = 0; r < 8; ++r) {
                int vr = vr0 + lhi * 8 + r;
                vu[((size_t)b * NO_ + vr) * OUT_ + h * DH_ + dt * 16 + l15] =
                    f2bf(acc2[dt][r]);
            }
    }
}

// ---------------------------------------------------------------------------
// v->q attention: per (b,h), S = q_qry . v_key^T over NO=256 keys, mask by
// v_mask, softmax, then q_update = P . v_val. 8 waves x 1 q-row tile.
// v_val B-fragments are gathered from global (L2-resident) to keep the
// workgroup's static LDS at 64KB.
// ---------------------------------------------------------------------------
__global__ __launch_bounds__(256) void attn_v2q(
        const uint16_t* __restrict__ vt, const uint16_t* __restrict__ qt,
        const float* __restrict__ vmask, uint16_t* __restrict__ qu) {
    const int h = blockIdx.x, b = blockIdx.y;
    const int tid = threadIdx.x, wave = tid >> 5, lane = tid & 31;
    const int l15 = lane & 15, lhi = lane >> 4;
    const float scale = 0.0883883476483184f;

    __shared__ uint16_t pbuf[8 * 16 * NO_];    // per-wave P scratch, 64KB
    uint16_t* pw = &pbuf[wave * 16 * NO_];

    float vm[16];
#pragma unroll
    for (int nt = 0; nt < 16; ++nt) vm[nt] = vmask[b * NO_ + nt * 16 + l15];

    const v8f vzero = {0.f,0.f,0.f,0.f,0.f,0.f,0.f,0.f};
    const int qr0 = wave * 16;

    v8f acc[16];
#pragma unroll
    for (int nt = 0; nt < 16; ++nt) acc[nt] = vzero;

    // ---- scores: S[16 x 256] ----
    for (int kk = 0; kk < 4; ++kk) {
        int k0 = kk * 32;
        frag32 fa;                              // A = q_qry
        {
            int qr = qr0 + l15, koff = lhi ? 16 : 0;
            const uint16_t* p =
                qt + ((size_t)b * ML_ + qr) * C6_ + OUT_ + h * DH_ + k0 + koff;
            fa.u[0] = *(const uint4*)p;
            fa.u[1] = *(const uint4*)(p + 8);
        }
#pragma unroll
        for (int nt = 0; nt < 16; ++nt) {       // B = v_key
            frag32 fb;
            int tok = nt * 16 + l15, kb = lhi ? 16 : 0;
            const uint16_t* p =
                vt + ((size_t)b * NO_ + tok) * C6_ + h * DH_ + k0 + kb;
            fb.u[0] = *(const uint4*)p;
            fb.u[1] = *(const uint4*)(p + 8);
            acc[nt] = wmma_bf16(fa.v, fb.v, acc[nt]);
        }
    }

    // ---- masked softmax over 256 cols ----
#pragma unroll
    for (int r = 0; r < 8; ++r) {
        float mx = -3.0e38f;
#pragma unroll
        for (int nt = 0; nt < 16; ++nt) {
            float s = acc[nt][r] * scale;
            s = (vm[nt] == 0.0f) ? -1.0e30f : s;
            acc[nt][r] = s;
            mx = fmaxf(mx, s);
        }
#pragma unroll
        for (int xm = 1; xm < 16; xm <<= 1)
            mx = fmaxf(mx, __shfl_xor(mx, xm, 32));
        float sum = 0.f;
#pragma unroll
        for (int nt = 0; nt < 16; ++nt) {
            float e = __expf(acc[nt][r] - mx);
            acc[nt][r] = e;
            sum += e;
        }
#pragma unroll
        for (int xm = 1; xm < 16; xm <<= 1)
            sum += __shfl_xor(sum, xm, 32);
        float inv = 1.0f / sum;
        int pr = lhi * 8 + r;
#pragma unroll
        for (int nt = 0; nt < 16; ++nt)
            pw[pr * NO_ + nt * 16 + l15] = f2bf(acc[nt][r] * inv);
    }

    // ---- q_update tile = P(16x256) . v_val(256x128) ----
    v8f acc2[8];
#pragma unroll
    for (int dt = 0; dt < 8; ++dt) acc2[dt] = vzero;
    for (int mm = 0; mm < 8; ++mm) {
        int m0 = mm * 32;
        frag32 fa;
        {
            int koff = lhi ? 16 : 0;
            const uint16_t* p = pw + l15 * NO_ + m0 + koff;
            fa.u[0] = *(const uint4*)p;
            fa.u[1] = *(const uint4*)(p + 8);
        }
#pragma unroll
        for (int dt = 0; dt < 8; ++dt) {
            h16frag fb;                         // B = v_val column gather
            int d = dt * 16 + l15, kb = lhi ? 16 : 0;
#pragma unroll
            for (int e = 0; e < 16; ++e) {
                int tok = m0 + kb + e;
                fb.h[e] = vt[((size_t)b * NO_ + tok) * C6_ +
                             2 * OUT_ + h * DH_ + d];
            }
            acc2[dt] = wmma_bf16(fa.v, fb.v, acc2[dt]);
        }
    }
#pragma unroll
    for (int dt = 0; dt < 8; ++dt)
#pragma unroll
        for (int r = 0; r < 8; ++r) {
            int qr = qr0 + lhi * 8 + r;
            qu[((size_t)b * ML_ + qr) * OUT_ + h * DH_ + dt * 16 + l15] =
                f2bf(acc2[dt][r]);
        }
}

// ---------------------------------------------------------------------------
// Host orchestration
// ---------------------------------------------------------------------------
extern "C" void kernel_launch(void* const* d_in, const int* in_sizes, int n_in,
                              void* d_out, int out_size, void* d_ws,
                              size_t ws_size, hipStream_t stream) {
    (void)in_sizes; (void)n_in; (void)out_size; (void)ws_size;

    const float* v    = (const float*)d_in[0];
    const float* q    = (const float*)d_in[1];
    const float* vmsk = (const float*)d_in[2];
    const float* qmsk = (const float*)d_in[3];
    const float* Wv   = (const float*)d_in[4];
    const float* bv   = (const float*)d_in[5];
    const float* Wq   = (const float*)d_in[6];
    const float* bq   = (const float*)d_in[7];
    const float* Wvo  = (const float*)d_in[8];
    const float* bvo  = (const float*)d_in[9];
    const float* Wqo  = (const float*)d_in[10];
    const float* bqo  = (const float*)d_in[11];

    float* out_v = (float*)d_out;                        // [64,256,2048]
    float* out_q = out_v + (size_t)NB * NO_ * OUT_;      // [64,128,2048]

    const size_t Mv = (size_t)NB * NO_;                  // 16384
    const size_t Mq = (size_t)NB * ML_;                  // 8192

    char* ws = (char*)d_ws;
    size_t off = 0;
    auto alloc = [&](size_t elems) {
        uint16_t* p = (uint16_t*)(ws + off);
        off += ((elems * 2 + 255) & ~(size_t)255);
        return p;
    };
    uint16_t* vb   = alloc(Mv * 2048);       // v in bf16
    uint16_t* qb   = alloc(Mq * 1024);       // q in bf16
    uint16_t* Wvb  = alloc((size_t)C6_ * 2048);
    uint16_t* Wqb  = alloc((size_t)C6_ * 1024);
    uint16_t* Wvob = alloc((size_t)OUT_ * 4096);
    uint16_t* Wqob = alloc((size_t)OUT_ * 3072);
    uint16_t* vt   = alloc(Mv * C6_);        // v_trans bf16
    uint16_t* qt   = alloc(Mq * C6_);        // q_trans bf16
    uint16_t* vu   = alloc(Mv * OUT_);       // v_update bf16
    uint16_t* qu   = alloc(Mq * OUT_);       // q_update bf16

    auto cvt = [&](const float* src, uint16_t* dst, size_t n) {
        cvt_f32_bf16<<<dim3((unsigned)(n / 1024)), 256, 0, stream>>>(
            src, dst, (long long)n);
    };
    cvt(v,   vb,   Mv * 2048);
    cvt(q,   qb,   Mq * 1024);
    cvt(Wv,  Wvb,  (size_t)C6_ * 2048);
    cvt(Wq,  Wqb,  (size_t)C6_ * 1024);
    cvt(Wvo, Wvob, (size_t)OUT_ * 4096);
    cvt(Wqo, Wqob, (size_t)OUT_ * 3072);

    // projections (+bias, +row mask), bf16 out
    gemm_wmma<true, true>
        <<<dim3(C6_ / 128, (unsigned)(Mv / 128)), 256, 0, stream>>>(
        vb, nullptr, 2048, Wvb, bv, vmsk, (int)Mv, C6_, 2048, vt, nullptr);
    gemm_wmma<true, true>
        <<<dim3(C6_ / 128, (unsigned)(Mq / 128)), 256, 0, stream>>>(
        qb, nullptr, 1024, Wqb, bq, qmsk, (int)Mq, C6_, 1024, qt, nullptr);

    // cross attention
    attn_q2v<<<dim3(16, NB), 256, 0, stream>>>(vt, qt, qmsk, vu);
    attn_v2q<<<dim3(16, NB), 256, 0, stream>>>(vt, qt, vmsk, qu);

    // output projections over concat(x, x_update), f32 out
    gemm_wmma<false, false>
        <<<dim3(OUT_ / 128, (unsigned)(Mv / 128)), 256, 0, stream>>>(
        vb, vu, 2048, Wvob, bvo, nullptr, (int)Mv, OUT_, 4096, nullptr, out_v);
    gemm_wmma<false, false>
        <<<dim3(OUT_ / 128, (unsigned)(Mq / 128)), 256, 0, stream>>>(
        qb, qu, 1024, Wqob, bqo, nullptr, (int)Mq, OUT_, 3072, nullptr, out_q);
}
